// BeamDecoder_40243843563777
// MI455X (gfx1250) — compile-verified, matrix operand order checked by reading, not run on hardware
//
#include <hip/hip_runtime.h>
#include <hip/hip_bf16.h>
#include <math.h>

// ---------------- problem constants ----------------
#define S_LEN   256
#define BSZ     32
#define HDIM    512
#define EDIM    256
#define VOCAB   10000
#define KBEAM   3
#define TSTEPS  32
#define BK      96          // B*K rows in beam phase
#define XK      1280        // concat [ex(256) | ctx(512) | h(512)]
#define GDIM    2048        // 4*H gates
#define BOS_TOK 1
#define EOS_TOK 1
#define NEGV    (-1000000000.0f)

typedef __bf16 bf16;
typedef __bf16 bf16x8  __attribute__((ext_vector_type(8)));
typedef __bf16 bf16x16 __attribute__((ext_vector_type(16)));
typedef float  v8f     __attribute__((ext_vector_type(8)));

__device__ __forceinline__ float sigmoidf_(float x) { return 1.0f / (1.0f + __expf(-x)); }

__device__ __forceinline__ bf16x16 join8(bf16x8 lo, bf16x8 hi) {
  return __builtin_shufflevector(lo, hi, 0,1,2,3,4,5,6,7,8,9,10,11,12,13,14,15);
}
__device__ __forceinline__ bf16x16 frag32(const bf16* p) {
  return join8(*(const bf16x8*)p, *(const bf16x8*)(p + 16));
}

// ---------------- prep kernels ----------------
__global__ void k_f32_to_bf16(const float* __restrict__ src, bf16* __restrict__ dst, int n) {
  for (int i = blockIdx.x * blockDim.x + threadIdx.x; i < n; i += gridDim.x * blockDim.x)
    dst[i] = (bf16)src[i];
}

// src is K x N row-major (fp32), dst is N x K row-major (bf16)
__global__ void k_transpose_bf16(const float* __restrict__ src, bf16* __restrict__ dst, int K, int N) {
  int total = K * N;
  for (int i = blockIdx.x * blockDim.x + threadIdx.x; i < total; i += gridDim.x * blockDim.x) {
    int k = i / N, n = i - k * N;
    dst[(size_t)n * K + k] = (bf16)src[i];
  }
}

// Combined gate weight, transposed: dst[n][k], k<768 from W_ih[k][n], else W_hh[k-768][n]
__global__ void k_build_WgT(const float* __restrict__ Wih, const float* __restrict__ Whh,
                            bf16* __restrict__ dst) {
  int total = GDIM * XK;
  for (int i = blockIdx.x * blockDim.x + threadIdx.x; i < total; i += gridDim.x * blockDim.x) {
    int n = i / XK, k = i - n * XK;
    float v = (k < 768) ? Wih[(size_t)k * GDIM + n] : Whh[(size_t)(k - 768) * GDIM + n];
    dst[i] = (bf16)v;
  }
}

// Wc transposed + rows remapped to X order [ex | ctx | h2]:
// X col k<256 -> Wc row k (ex); 256<=k<768 -> Wc row k+512 (ctx); k>=768 -> Wc row k-512 (h2)
__global__ void k_build_WcT(const float* __restrict__ Wc, bf16* __restrict__ dst) {
  size_t total = (size_t)VOCAB * XK;
  for (size_t i = blockIdx.x * blockDim.x + threadIdx.x; i < total; i += (size_t)gridDim.x * blockDim.x) {
    int n = (int)(i / XK), k = (int)(i - (size_t)n * XK);
    int r = (k < 256) ? k : (k < 768) ? (k + 512) : (k - 512);
    dst[i] = (bf16)Wc[(size_t)r * VOCAB + n];
  }
}

__global__ void k_bias_sum(const float* a, const float* b, float* out, int n) {
  int i = blockIdx.x * blockDim.x + threadIdx.x;
  if (i < n) out[i] = a[i] + b[i];
}

__global__ void k_init(int* tok, int* p0, int* p1) {
  int i = blockIdx.x * blockDim.x + threadIdx.x;
  if (i < TSTEPS * BK) { p0[i] = 0; p1[i] = 0; }
  if (i < BK) tok[i] = BOS_TOK;
}

__global__ void k_h0_to_X(const float* __restrict__ hTmp, bf16* __restrict__ X) {
  int row = blockIdx.x, h = blockIdx.y * 256 + threadIdx.x;
  X[(size_t)row * XK + 768 + h] = (bf16)hTmp[row * HDIM + h];
}

// ---------------- WMMA bf16 GEMM: C[M,N] = A[M,K] * BT[N,K]^T (+bias) ----------------
// Per wave: 16(M) x (16*JT)(N) tile; 4 waves stacked in M per block.
// JT is a compile-time tile count -> branch-free straight-line inner loop.
// All fragments of a k-chunk are loaded into distinct registers before the WMMA
// group so the loads can be clause-batched and covered by a single wait.
template <int JT>
__global__ __launch_bounds__(128) void k_gemm_bf16(
    const bf16* __restrict__ A, int lda,
    const bf16* __restrict__ BT,            // N x K row-major
    const float* __restrict__ bias,         // may be null
    float* __restrict__ C, int ldc,
    int M, int Kd, int nbase)
{
  const int lane = threadIdx.x & 31;
  const int wave = threadIdx.x >> 5;
  const int half = lane >> 4;       // 0 or 1
  const int mr   = lane & 15;       // row-within-tile (A) / col-within-tile (B,C)
  const int mtile  = blockIdx.y * 64 + wave * 16;
  const int ntile0 = nbase + blockIdx.x * (16 * JT);
  if (mtile >= M) return;

  v8f acc0{}, acc1{}, acc2{}, acc3{};

  const bf16* arow = A + (size_t)(mtile + mr) * lda + half * 8;
  const bf16* brow0 =                 BT + (size_t)(ntile0 +  0 + mr) * Kd + half * 8;
  const bf16* brow1 = (JT > 1) ? BT + (size_t)(ntile0 + 16 + mr) * Kd + half * 8 : brow0;
  const bf16* brow2 = (JT > 2) ? BT + (size_t)(ntile0 + 32 + mr) * Kd + half * 8 : brow0;
  const bf16* brow3 = (JT > 3) ? BT + (size_t)(ntile0 + 48 + mr) * Kd + half * 8 : brow0;

#pragma unroll 2
  for (int k0 = 0; k0 < Kd; k0 += 32) {
    __builtin_prefetch(arow + k0 + 64, 0, 3);
    __builtin_prefetch(brow0 + k0 + 64, 0, 3);
    if constexpr (JT > 1) __builtin_prefetch(brow1 + k0 + 64, 0, 3);
    if constexpr (JT > 2) __builtin_prefetch(brow2 + k0 + 64, 0, 3);
    if constexpr (JT > 3) __builtin_prefetch(brow3 + k0 + 64, 0, 3);

    // load all fragments for this k-chunk up front (distinct registers)
    bf16x16 af  = frag32(arow + k0);
    bf16x16 bf0 = frag32(brow0 + k0);
    bf16x16 bf1{}, bf2{}, bf3{};
    if constexpr (JT > 1) bf1 = frag32(brow1 + k0);
    if constexpr (JT > 2) bf2 = frag32(brow2 + k0);
    if constexpr (JT > 3) bf3 = frag32(brow3 + k0);

    // back-to-back WMMAs on the staged fragments
    acc0 = __builtin_amdgcn_wmma_f32_16x16x32_bf16(false, af, false, bf0, (short)0, acc0, false, false);
    if constexpr (JT > 1)
      acc1 = __builtin_amdgcn_wmma_f32_16x16x32_bf16(false, af, false, bf1, (short)0, acc1, false, false);
    if constexpr (JT > 2)
      acc2 = __builtin_amdgcn_wmma_f32_16x16x32_bf16(false, af, false, bf2, (short)0, acc2, false, false);
    if constexpr (JT > 3)
      acc3 = __builtin_amdgcn_wmma_f32_16x16x32_bf16(false, af, false, bf3, (short)0, acc3, false, false);
  }

  // C layout: VGPR r -> M = half*8 + r, N = mr (within 16x16 tile)
  const int crow = mtile + half * 8;
#pragma unroll
  for (int j = 0; j < JT; ++j) {
    int col = ntile0 + j * 16 + mr;
    float bv = bias ? bias[col] : 0.0f;
    const v8f& acc = (j == 0) ? acc0 : (j == 1) ? acc1 : (j == 2) ? acc2 : acc3;
#pragma unroll
    for (int r = 0; r < 8; ++r) {
      C[(size_t)(crow + r) * ldc + col] = acc[r] + bv;
    }
  }
}

// ---------------- decode-step kernels ----------------
// ex slot of X from embedding
__global__ void k_embed(const int* __restrict__ tok, const bf16* __restrict__ embB,
                        bf16* __restrict__ X) {
  int row = blockIdx.x, e = threadIdx.x;   // 256 threads
  X[(size_t)row * XK + e] = embB[(size_t)tok[row] * EDIM + e];
}

// sc[row][s] = sum_h tanh(pp[row][h] + encp[s*B+b][h]) * Wv[h]  (masked)
__global__ __launch_bounds__(256) void k_attn_scores(
    const float* __restrict__ pp, const float* __restrict__ encp,
    const float* __restrict__ Wv, const float* __restrict__ mask,
    float* __restrict__ sc, int kdiv)
{
  int wave = threadIdx.x >> 5, lane = threadIdx.x & 31;
  int s = blockIdx.x * 8 + wave;
  int row = blockIdx.y;
  int b = row / kdiv;
  const float* pr = pp + (size_t)row * HDIM;
  const float* er = encp + ((size_t)s * BSZ + b) * HDIM;
  float acc = 0.0f;
  for (int h = lane; h < HDIM; h += 32)
    acc += tanhf(pr[h] + er[h]) * Wv[h];
#pragma unroll
  for (int off = 16; off > 0; off >>= 1) acc += __shfl_xor(acc, off, 32);
  if (lane == 0) {
    float m = mask[b * S_LEN + s];
    sc[(size_t)row * S_LEN + s] = (m == 0.0f) ? NEGV : acc;
  }
}

__global__ __launch_bounds__(256) void k_softmax_s(float* __restrict__ sc) {
  int row = blockIdx.x, t = threadIdx.x;            // S_LEN == 256 threads
  __shared__ float red[256];
  float v = sc[(size_t)row * S_LEN + t];
  red[t] = v; __syncthreads();
  for (int o = 128; o > 0; o >>= 1) { if (t < o) red[t] = fmaxf(red[t], red[t + o]); __syncthreads(); }
  float mx = red[0]; __syncthreads();
  float e = __expf(v - mx);
  red[t] = e; __syncthreads();
  for (int o = 128; o > 0; o >>= 1) { if (t < o) red[t] += red[t + o]; __syncthreads(); }
  sc[(size_t)row * S_LEN + t] = e / red[0];
}

// ctx[row][h] = sum_s a[row][s] * enc[s][b][h]; write as bf16 into X ctx slot
__global__ __launch_bounds__(256) void k_context(
    const float* __restrict__ a, const float* __restrict__ enc,
    bf16* __restrict__ X, int kdiv)
{
  int row = blockIdx.x, h = blockIdx.y * 256 + threadIdx.x;
  int b = row / kdiv;
  __shared__ float aw[S_LEN];
  aw[threadIdx.x] = a[(size_t)row * S_LEN + threadIdx.x];
  __syncthreads();
  float acc = 0.0f;
  for (int s = 0; s < S_LEN; ++s)
    acc += aw[s] * enc[((size_t)s * BSZ + b) * HDIM + h];
  X[(size_t)row * XK + 256 + h] = (bf16)acc;
}

__global__ __launch_bounds__(256) void k_lstm(
    const float* __restrict__ g, const float* __restrict__ c,
    float* __restrict__ c2, bf16* __restrict__ h2B, bf16* __restrict__ X)
{
  int row = blockIdx.x, h = blockIdx.y * 256 + threadIdx.x;
  const float* gr = g + (size_t)row * GDIM;
  float ii = sigmoidf_(gr[h]);
  float ff = sigmoidf_(gr[HDIM + h]);
  float gg = tanhf(gr[2 * HDIM + h]);
  float oo = sigmoidf_(gr[3 * HDIM + h]);
  float cc = ff * c[(size_t)row * HDIM + h] + ii * gg;
  float hh = oo * tanhf(cc);
  c2[(size_t)row * HDIM + h] = cc;
  h2B[(size_t)row * HDIM + h] = (bf16)hh;
  X[(size_t)row * XK + 768 + h] = (bf16)hh;
}

__global__ __launch_bounds__(1024) void k_logsoftmax(float* __restrict__ lp) {
  int row = blockIdx.x, t = threadIdx.x;
  float* rp = lp + (size_t)row * VOCAB;
  float loc[10];
  float m = -1e30f;
#pragma unroll
  for (int i = 0; i < 10; ++i) {
    int v = t + i * 1024;
    loc[i] = (v < VOCAB) ? rp[v] : -1e30f;
    m = fmaxf(m, loc[i]);
  }
  __shared__ float red[1024];
  red[t] = m; __syncthreads();
  for (int o = 512; o > 0; o >>= 1) { if (t < o) red[t] = fmaxf(red[t], red[t + o]); __syncthreads(); }
  float mx = red[0]; __syncthreads();
  float s = 0.0f;
#pragma unroll
  for (int i = 0; i < 10; ++i) {
    int v = t + i * 1024;
    if (v < VOCAB) s += __expf(loc[i] - mx);
  }
  red[t] = s; __syncthreads();
  for (int o = 512; o > 0; o >>= 1) { if (t < o) red[t] += red[t + o]; __syncthreads(); }
  float lse = mx + __logf(red[0]);
#pragma unroll
  for (int i = 0; i < 10; ++i) {
    int v = t + i * 1024;
    if (v < VOCAB) rp[v] = loc[i] - lse;
  }
}

__device__ __forceinline__ void top3_insert(float x, int ix,
    float& v0, int& i0, float& v1, int& i1, float& v2, int& i2) {
  if (x > v0 || (x == v0 && ix < i0)) { v2 = v1; i2 = i1; v1 = v0; i1 = i0; v0 = x; i0 = ix; }
  else if (x > v1 || (x == v1 && ix < i1)) { v2 = v1; i2 = i1; v1 = x; i1 = ix; }
  else if (x > v2 || (x == v2 && ix < i2)) { v2 = x; i2 = ix; }
}

// step-0 top-k: per batch b, top-3 over V of lp[b][:]
__global__ __launch_bounds__(256) void k_topk_first(
    const float* __restrict__ lp, int* tok, float* cum, int* eos, int* gidx, int* preds)
{
  int b = blockIdx.x, t = threadIdx.x;
  const float* rp = lp + (size_t)b * VOCAB;
  float v0 = -1e30f, v1 = -1e30f, v2 = -1e30f; int i0 = 0, i1 = 0, i2 = 0;
  for (int v = t; v < VOCAB; v += 256) top3_insert(rp[v], v, v0, i0, v1, i1, v2, i2);
  __shared__ float sv[768]; __shared__ int si[768];
  sv[t] = v0; si[t] = i0; sv[256 + t] = v1; si[256 + t] = i1; sv[512 + t] = v2; si[512 + t] = i2;
  __syncthreads();
  if (t == 0) {
    float b0 = -1e30f, b1 = -1e30f, b2 = -1e30f; int j0 = 0, j1 = 0, j2 = 0;
    for (int q = 0; q < 768; ++q) top3_insert(sv[q], si[q], b0, j0, b1, j1, b2, j2);
    float bv[3] = {b0, b1, b2}; int bi[3] = {j0, j1, j2};
    for (int j = 0; j < 3; ++j) {
      int row = b * KBEAM + j;
      tok[row] = bi[j]; cum[row] = bv[j];
      eos[row] = (bi[j] == EOS_TOK) ? 1 : 0;
      gidx[row] = b;                    // repeat h2/c2 across beams
      preds[row] = bi[j];               // preds[0][row]
    }
  }
}

// beam top-k: per batch b, top-3 over K*V of cum + stp
__global__ __launch_bounds__(256) void k_topk_beam(
    const float* __restrict__ lp, const float* __restrict__ cum, const int* __restrict__ eos,
    int* tokN, float* cumN, int* eosN, int* gidx)
{
  int b = blockIdx.x, t = threadIdx.x;
  float v0 = -1e30f, v1 = -1e30f, v2 = -1e30f; int i0 = 0, i1 = 0, i2 = 0;
  for (int v = t; v < KBEAM * VOCAB; v += 256) {
    int k = v / VOCAB, vv = v - k * VOCAB;
    int r = b * KBEAM + k;
    float x;
    if (eos[r]) x = cum[r] + ((vv == EOS_TOK) ? 0.0f : NEGV);
    else        x = cum[r] + lp[(size_t)r * VOCAB + vv];
    top3_insert(x, v, v0, i0, v1, i1, v2, i2);
  }
  __shared__ float sv[768]; __shared__ int si[768];
  sv[t] = v0; si[t] = i0; sv[256 + t] = v1; si[256 + t] = i1; sv[512 + t] = v2; si[512 + t] = i2;
  __syncthreads();
  if (t == 0) {
    float b0 = -1e30f, b1 = -1e30f, b2 = -1e30f; int j0 = 0, j1 = 0, j2 = 0;
    for (int q = 0; q < 768; ++q) top3_insert(sv[q], si[q], b0, j0, b1, j1, b2, j2);
    float bv[3] = {b0, b1, b2}; int bi[3] = {j0, j1, j2};
    for (int j = 0; j < 3; ++j) {
      int row = b * KBEAM + j;
      int beam = bi[j] / VOCAB, ntok = bi[j] - beam * VOCAB;
      int gs = b * KBEAM + beam;
      gidx[row] = gs; tokN[row] = ntok; cumN[row] = bv[j];
      eosN[row] = (eos[gs] || ntok == EOS_TOK) ? 1 : 0;
    }
  }
}

__global__ void k_commit(int* tok, float* cum, int* eos,
                         const int* tokN, const float* cumN, const int* eosN) {
  int i = threadIdx.x;
  if (i < BK) { tok[i] = tokN[i]; cum[i] = cumN[i]; eos[i] = eosN[i]; }
}

__global__ __launch_bounds__(256) void k_gather(
    const int* __restrict__ gidx, const float* __restrict__ c2, float* __restrict__ c,
    const bf16* __restrict__ h2B, bf16* __restrict__ X)
{
  int row = blockIdx.x, h = blockIdx.y * 256 + threadIdx.x;
  int gs = gidx[row];
  c[(size_t)row * HDIM + h] = c2[(size_t)gs * HDIM + h];
  X[(size_t)row * XK + 768 + h] = h2B[(size_t)gs * HDIM + h];
}

__global__ void k_preds_perm(const int* __restrict__ pold, int* __restrict__ pnew,
                             const int* __restrict__ gidx, const int* __restrict__ tokN, int tstep) {
  int tt = blockIdx.x, r = threadIdx.x;   // 32 blocks x 96 threads
  pnew[tt * BK + r] = (tt == tstep) ? tokN[r] : pold[tt * BK + gidx[r]];
}

__global__ void k_output(const int* __restrict__ preds, const float* __restrict__ cum,
                         float* __restrict__ out, int out_size) {
  int i = blockIdx.x * blockDim.x + threadIdx.x;
  if (i >= out_size) return;
  if (i < TSTEPS * BSZ) {
    int t = i / BSZ, b = i - t * BSZ;
    out[i] = (float)preds[t * BK + b * KBEAM];   // preds[:, ::K]
  } else if (i < TSTEPS * BSZ + BSZ * KBEAM) {
    out[i] = cum[i - TSTEPS * BSZ];              // best_scores (B,K) row-major
  }
}

// ---------------- host orchestration ----------------
extern "C" void kernel_launch(void* const* d_in, const int* in_sizes, int n_in,
                              void* d_out, int out_size, void* d_ws, size_t ws_size,
                              hipStream_t stream) {
  (void)in_sizes; (void)n_in; (void)ws_size;
  const float* enc    = (const float*)d_in[0];
  const float* last_h = (const float*)d_in[1];
  const float* last_c = (const float*)d_in[2];
  const float* amask  = (const float*)d_in[3];
  // d_in[4] = indices (unused by reference)
  const float* emb    = (const float*)d_in[5];
  const float* Wp     = (const float*)d_in[6];
  const float* We     = (const float*)d_in[7];
  const float* Wv     = (const float*)d_in[8];
  const float* Wih    = (const float*)d_in[9];
  const float* Whh    = (const float*)d_in[10];
  const float* bih    = (const float*)d_in[11];
  const float* bhh    = (const float*)d_in[12];
  const float* Wc     = (const float*)d_in[13];
  const float* bc     = (const float*)d_in[14];
  const float* Winit  = (const float*)d_in[15];
  const float* binit  = (const float*)d_in[16];

  // workspace carve-out (256B aligned)
  char* w = (char*)d_ws; size_t off = 0;
  auto alloc = [&](size_t bytes) -> char* {
    char* p = w + off; off = (off + bytes + 255) & ~(size_t)255; return p;
  };
  bf16*  encB   = (bf16*)alloc((size_t)S_LEN * BSZ * HDIM * 2);
  bf16*  WeT    = (bf16*)alloc((size_t)HDIM * HDIM * 2);
  bf16*  WpT    = (bf16*)alloc((size_t)HDIM * HDIM * 2);
  bf16*  WinT   = (bf16*)alloc((size_t)HDIM * HDIM * 2);
  bf16*  WgT    = (bf16*)alloc((size_t)GDIM * XK * 2);
  bf16*  WcT    = (bf16*)alloc((size_t)VOCAB * XK * 2);
  bf16*  embB   = (bf16*)alloc((size_t)VOCAB * EDIM * 2);
  bf16*  lastHB = (bf16*)alloc((size_t)BSZ * HDIM * 2);
  bf16*  lastCB = (bf16*)alloc((size_t)BSZ * HDIM * 2);
  float* biasG  = (float*)alloc((size_t)GDIM * 4);
  float* encp   = (float*)alloc((size_t)S_LEN * BSZ * HDIM * 4);
  bf16*  X      = (bf16*)alloc((size_t)BK * XK * 2);
  float* pp     = (float*)alloc((size_t)BK * HDIM * 4);
  float* sc     = (float*)alloc((size_t)BK * S_LEN * 4);
  float* gbuf   = (float*)alloc((size_t)BK * GDIM * 4);
  float* lp     = (float*)alloc((size_t)BK * VOCAB * 4);
  float* cbuf   = (float*)alloc((size_t)BK * HDIM * 4);
  float* c2     = (float*)alloc((size_t)BK * HDIM * 4);
  bf16*  h2B    = (bf16*)alloc((size_t)BK * HDIM * 2);
  float* hTmp   = (float*)alloc((size_t)BK * HDIM * 4);
  int*   tok    = (int*)alloc(BK * 4);
  float* cum    = (float*)alloc(BK * 4);
  int*   eos    = (int*)alloc(BK * 4);
  int*   gidx   = (int*)alloc(BK * 4);
  int*   tokN   = (int*)alloc(BK * 4);
  float* cumN   = (float*)alloc(BK * 4);
  int*   eosN   = (int*)alloc(BK * 4);
  int*   preds0 = (int*)alloc((size_t)TSTEPS * BK * 4);
  int*   preds1 = (int*)alloc((size_t)TSTEPS * BK * 4);

  auto gemm = [&](const bf16* A, int lda, const bf16* BT, const float* bias,
                  float* C, int ldc, int M, int N, int Kd) {
    int my = (M + 63) / 64;
    int nfull = N & ~63;
    if (nfull > 0)
      k_gemm_bf16<4><<<dim3(nfull / 64, my), 128, 0, stream>>>(A, lda, BT, bias, C, ldc, M, Kd, 0);
    int rem = N - nfull;             // multiple of 16 for all our shapes
    if (rem == 16)
      k_gemm_bf16<1><<<dim3(1, my), 128, 0, stream>>>(A, lda, BT, bias, C, ldc, M, Kd, nfull);
    else if (rem == 32)
      k_gemm_bf16<2><<<dim3(1, my), 128, 0, stream>>>(A, lda, BT, bias, C, ldc, M, Kd, nfull);
    else if (rem == 48)
      k_gemm_bf16<3><<<dim3(1, my), 128, 0, stream>>>(A, lda, BT, bias, C, ldc, M, Kd, nfull);
  };

  // ---- prep: weight conversion / reordering (deterministic every call) ----
  k_f32_to_bf16<<<1024, 256, 0, stream>>>(enc, encB, S_LEN * BSZ * HDIM);
  k_transpose_bf16<<<512, 256, 0, stream>>>(We, WeT, HDIM, HDIM);
  k_transpose_bf16<<<512, 256, 0, stream>>>(Wp, WpT, HDIM, HDIM);
  k_transpose_bf16<<<512, 256, 0, stream>>>(Winit, WinT, HDIM, HDIM);
  k_build_WgT<<<2048, 256, 0, stream>>>(Wih, Whh, WgT);
  k_build_WcT<<<4096, 256, 0, stream>>>(Wc, WcT);
  k_f32_to_bf16<<<2048, 256, 0, stream>>>(emb, embB, VOCAB * EDIM);
  k_f32_to_bf16<<<64, 256, 0, stream>>>(last_h + BSZ * HDIM, lastHB, BSZ * HDIM); // last_h[-1]
  k_f32_to_bf16<<<64, 256, 0, stream>>>(last_c + BSZ * HDIM, lastCB, BSZ * HDIM);
  k_bias_sum<<<(GDIM + 255) / 256, 256, 0, stream>>>(bih, bhh, biasG, GDIM);

  // enc_proj0 = encoder_states @ We   (8192 x 512 x 512)
  gemm(encB, HDIM, WeT, nullptr, encp, HDIM, S_LEN * BSZ, HDIM, HDIM);
  // h0 = last_h[-1] @ W_init + b_init ; c0 likewise
  gemm(lastHB, HDIM, WinT, binit, hTmp, HDIM, BSZ, HDIM, HDIM);
  gemm(lastCB, HDIM, WinT, binit, cbuf, HDIM, BSZ, HDIM, HDIM);
  k_h0_to_X<<<dim3(BSZ, 2), 256, 0, stream>>>(hTmp, X);
  k_init<<<(TSTEPS * BK + 255) / 256, 256, 0, stream>>>(tok, preds0, preds1);

  int* pcur = preds0; int* pnext = preds1;

  auto decode_core = [&](int R, int kdiv) {
    k_embed<<<R, EDIM, 0, stream>>>(tok, embB, X);
    // pp = h @ Wp : A is the h-slot of X (lda = XK)
    gemm(X + 768, XK, WpT, nullptr, pp, HDIM, R, HDIM, HDIM);
    k_attn_scores<<<dim3(S_LEN / 8, R), 256, 0, stream>>>(pp, encp, Wv, amask, sc, kdiv);
    k_softmax_s<<<R, 256, 0, stream>>>(sc);
    k_context<<<dim3(R, 2), 256, 0, stream>>>(sc, enc, X, kdiv);
    // gates = [ex|ctx|h] @ [W_ih;W_hh] + (b_ih+b_hh)
    gemm(X, XK, WgT, biasG, gbuf, GDIM, R, GDIM, XK);
    k_lstm<<<dim3(R, 2), 256, 0, stream>>>(gbuf, cbuf, c2, h2B, X);
    // logits = [ex|ctx|h2] @ Wc' + bc   (Wc rows pre-reordered to X layout)
    gemm(X, XK, WcT, bc, lp, VOCAB, R, VOCAB, XK);
    k_logsoftmax<<<R, 1024, 0, stream>>>(lp);
  };

  // ---- step 0 (R = 32, b = row) ----
  decode_core(BSZ, 1);
  k_topk_first<<<BSZ, 256, 0, stream>>>(lp, tok, cum, eos, gidx, pcur);
  k_gather<<<dim3(BK, 2), 256, 0, stream>>>(gidx, c2, cbuf, h2B, X);

  // ---- steps 1..T-1 (R = 96, b = row / K) ----
  for (int t = 1; t < TSTEPS; ++t) {
    decode_core(BK, KBEAM);
    k_topk_beam<<<BSZ, 256, 0, stream>>>(lp, cum, eos, tokN, cumN, eosN, gidx);
    k_preds_perm<<<TSTEPS, BK, 0, stream>>>(pcur, pnext, gidx, tokN, t);
    { int* tmp = pcur; pcur = pnext; pnext = tmp; }
    k_commit<<<1, 128, 0, stream>>>(tok, cum, eos, tokN, cumN, eosN);
    k_gather<<<dim3(BK, 2), 256, 0, stream>>>(gidx, c2, cbuf, h2B, X);
  }

  k_output<<<(out_size + 255) / 256, 256, 0, stream>>>(pcur, cum, (float*)d_out, out_size);
}
